// FrameletLayer_4174708211748
// MI455X (gfx1250) — matrix operand order repaired and checked by reading, not compiled, not run on hardware
//
#include <hip/hip_runtime.h>
#include <hip/hip_bf16.h>

#define C 128

typedef __attribute__((ext_vector_type(16))) __bf16 v16bf;
typedef __attribute__((ext_vector_type(8)))  float  v8f;

union BfChunk { unsigned u[8]; uint4 q[2]; v16bf v; };

// ---------- helpers ----------

// A-matrix 16x32 bf16 per-lane pack (ISA 7.12.2): lane holds row m=lane&15,
// K-runs [kb..kb+7] and [kb+16..kb+23] where kb = 8*(lane>>4) within the 32-chunk.
// Native f32->bf16 casts lower to v_cvt_pk_bf16_f32.
__device__ inline v16bf packA(const float* __restrict__ p) {
    v16bf v;
#pragma unroll
    for (int q = 0; q < 8; ++q) v[q]     = (__bf16)p[q];
#pragma unroll
    for (int q = 0; q < 8; ++q) v[8 + q] = (__bf16)p[16 + q];
    return v;
}
__device__ inline v16bf negbf(v16bf b) {
    BfChunk w; w.v = b;
#pragma unroll
    for (int q = 0; q < 8; ++q) w.u[q] ^= 0x80008000u;
    return w.v;
}
// SiLU with fast reciprocal: denominator in (1,inf) so v_rcp_f32 is safe (~1 ulp).
__device__ inline float silu_fast(float v) {
    return v * __builtin_amdgcn_rcpf(1.f + __expf(-v));
}

// ---------- kernels ----------

// c[k] = sum_f theta[f] * approx[f][k]   (F=4, K=6)
__global__ void coeff_k(const float* __restrict__ theta, const float* __restrict__ approx,
                        float* __restrict__ c) {
    int k = threadIdx.x;
    if (k < 6) {
        float s = 0.f;
#pragma unroll
        for (int f = 0; f < 4; ++f) s += theta[f] * approx[f * 6 + k];
        c[k] = s;
    }
}

// acc = c0*x ; Tprev = x ; Tcur = 0
__global__ void init_k(const float* __restrict__ xr, const float* __restrict__ xi,
                       float* __restrict__ tpr, float* __restrict__ tpi,
                       float* __restrict__ tcr, float* __restrict__ tci,
                       float* __restrict__ ar,  float* __restrict__ ai,
                       const float* __restrict__ c, int n) {
    int i = blockIdx.x * blockDim.x + threadIdx.x;
    if (i >= n) return;
    float c0 = c[0];
    float r = xr[i], im = xi[i];
    tpr[i] = r;  tpi[i] = im;
    tcr[i] = 0.f; tci[i] = 0.f;
    ar[i] = c0 * r; ai[i] = c0 * im;
}

// acc += c[k]*Tcur ; optionally Tprev = -Tprev (init for next atomic pass)
__global__ void fuse_k(float* __restrict__ ar, float* __restrict__ ai,
                       const float* __restrict__ tcr, const float* __restrict__ tci,
                       float* __restrict__ tpr, float* __restrict__ tpi,
                       const float* __restrict__ c, int k, int prep, int n) {
    int i = blockIdx.x * blockDim.x + threadIdx.x;
    if (i >= n) return;
    float ck = c[k];
    ar[i] += ck * tcr[i];
    ai[i] += ck * tci[i];
    if (prep) { tpr[i] = -tpr[i]; tpi[i] = -tpi[i]; }
}

// One wave per edge: dst[row] += scale*w * src[col]  (real & imag fused).
// Lane-contiguous float4 gathers + per-channel fp32 atomics (L2-resident).
__global__ __launch_bounds__(256)
void edge_prop(const int* __restrict__ ei, const float* __restrict__ ew,
               const float* __restrict__ sr, const float* __restrict__ si,
               float* __restrict__ dr, float* __restrict__ di,
               float scale, int E) {
    int wid  = (blockIdx.x * blockDim.x + threadIdx.x) >> 5;
    int lane = threadIdx.x & 31;
    if (wid >= E) return;
    int row = ei[wid];
    int col = ei[E + wid];
    float f = scale * ew[wid];
    int so  = col * C + lane * 4;
    int dof = row * C + lane * 4;
    float4 vr = *(const float4*)(sr + so);
    float4 vi = *(const float4*)(si + so);
    unsafeAtomicAdd(dr + dof + 0, f * vr.x);
    unsafeAtomicAdd(dr + dof + 1, f * vr.y);
    unsafeAtomicAdd(dr + dof + 2, f * vr.z);
    unsafeAtomicAdd(dr + dof + 3, f * vr.w);
    unsafeAtomicAdd(di + dof + 0, f * vi.x);
    unsafeAtomicAdd(di + dof + 1, f * vi.y);
    unsafeAtomicAdd(di + dof + 2, f * vi.z);
    unsafeAtomicAdd(di + dof + 3, f * vi.w);
}

// yr = accR@Wr^T - accI@Wi^T ; yi = accR@Wi^T + accI@Wr^T ; SiLU both.
// Wr/Wi staged in LDS (64 KB) in exact WMMA-B per-lane layout (bf16).
// 8 waves/block, one 16x128 output row-tile per wave, v_wmma_f32_16x16x32_bf16.
// Sign handled by negating the A_imag operand once per tile (not B per step).
__global__ __launch_bounds__(256)
void gemm_silu(const float* __restrict__ accR, const float* __restrict__ accI,
               const float* __restrict__ Wr, const float* __restrict__ Wi,
               float* __restrict__ out, int N) {
    __shared__ uint4 ldsB[4096];     // [comp(2)][t(8)][s(4)][lane(32)] x 2 uint4 = 64 KB
    const int tid = threadIdx.x;

    // Stage: B[k][j] = W[j][k]; lane holds column j=t*16+(L&15),
    // K-runs [kb..kb+7] and [kb+16..kb+23], kb = s*32 + 8*(L>>4) (matches packA).
#pragma unroll
    for (int it = 0; it < 8; ++it) {
        int slot = it * 256 + tid;
        int L = slot & 31, s = (slot >> 5) & 3, t = (slot >> 7) & 7, comp = slot >> 10;
        int j  = t * 16 + (L & 15);
        int kb = s * 32 + 8 * (L >> 4);
        const float* wp = (comp ? Wi : Wr) + j * C + kb;
        BfChunk w;
#pragma unroll
        for (int q = 0; q < 16; ++q) w.v[q] = (__bf16)wp[q < 8 ? q : q + 8];
        int idx = ((((comp * 8 + t) * 4 + s) * 32) + L) * 2;
        ldsB[idx]     = w.q[0];
        ldsB[idx + 1] = w.q[1];
    }
    __syncthreads();

    int tile = blockIdx.x * 8 + (tid >> 5);
    int ntiles = N >> 4;                       // N=50000 -> 3125 full tiles
    if (tile >= ntiles) return;
    int lane = tid & 31;
    int kh   = lane >> 4;
    int mrow = tile * 16 + (lane & 15);

    // Load + pack A (real & imag), 16x128 each, K split into 4 chunks of 32.
    // nai = -a_imag (one-time sign flip replaces per-step B negation).
    v16bf arv[4], aiv[4], nai[4];
#pragma unroll
    for (int s = 0; s < 4; ++s) {
        int kb = s * 32 + 8 * kh;
        arv[s] = packA(accR + (size_t)mrow * C + kb);
        aiv[s] = packA(accI + (size_t)mrow * C + kb);
        nai[s] = negbf(aiv[s]);
    }

    size_t plane = (size_t)N * C;
#pragma unroll
    for (int t = 0; t < 8; ++t) {
        v8f cr = {}; v8f ci = {};
#pragma unroll
        for (int s = 0; s < 4; ++s) {
            int bidx = ((t * 4 + s) * 32 + lane) * 2;
            BfChunk wr2, wi2;
            wr2.q[0] = ldsB[bidx];        wr2.q[1] = ldsB[bidx + 1];
            wi2.q[0] = ldsB[bidx + 2048]; wi2.q[1] = ldsB[bidx + 2049];
            v16bf br = wr2.v;
            v16bf bi = wi2.v;
            cr = __builtin_amdgcn_wmma_f32_16x16x32_bf16(false, arv[s], false, br, (short)0, cr, false, false);
            cr = __builtin_amdgcn_wmma_f32_16x16x32_bf16(false, nai[s], false, bi, (short)0, cr, false, false);
            ci = __builtin_amdgcn_wmma_f32_16x16x32_bf16(false, arv[s], false, bi, (short)0, ci, false, false);
            ci = __builtin_amdgcn_wmma_f32_16x16x32_bf16(false, aiv[s], false, br, (short)0, ci, false, false);
        }
        // C/D layout: col = lane&15, row-in-tile = g + 8*(lane>>4)
        int ncol = t * 16 + (lane & 15);
#pragma unroll
        for (int g = 0; g < 8; ++g) {
            int r = tile * 16 + g + 8 * kh;
            out[(size_t)r * C + ncol]         = silu_fast(cr[g]);
            out[plane + (size_t)r * C + ncol] = silu_fast(ci[g]);
        }
    }
}

// ---------- launch ----------
extern "C" void kernel_launch(void* const* d_in, const int* in_sizes, int n_in,
                              void* d_out, int out_size, void* d_ws, size_t ws_size,
                              hipStream_t stream) {
    const float* xr     = (const float*)d_in[0];
    const float* xi     = (const float*)d_in[1];
    const int*   ei     = (const int*)  d_in[2];
    const float* ew     = (const float*)d_in[3];
    const float* approx = (const float*)d_in[4];
    const float* theta  = (const float*)d_in[5];
    const float* Wr     = (const float*)d_in[6];
    const float* Wi     = (const float*)d_in[7];

    int NC = in_sizes[0];        // N*C
    int N  = NC / C;
    int E  = in_sizes[3];

    float* ws  = (float*)d_ws;
    float* c   = ws;             // 6 coeffs (padded to 64)
    float* b0  = ws + 64;
    float* tpr = b0;
    float* tpi = b0 + (size_t)NC;
    float* tcr = b0 + 2 * (size_t)NC;
    float* tci = b0 + 3 * (size_t)NC;
    float* ar  = b0 + 4 * (size_t)NC;
    float* ai  = b0 + 5 * (size_t)NC;

    coeff_k<<<1, 32, 0, stream>>>(theta, approx, c);

    int ewb = (NC + 255) / 256;
    init_k<<<ewb, 256, 0, stream>>>(xr, xi, tpr, tpi, tcr, tci, ar, ai, c, NC);

    int eb = (E + 7) / 8;        // one wave (32 lanes) per edge, 8 waves/block
    // T1 = -A(T0): Tcur (zeroed) += -1*w*Tprev[col]
    edge_prop<<<eb, 256, 0, stream>>>(ei, ew, tpr, tpi, tcr, tci, -1.0f, E);

    for (int ki = 2; ki < 6; ++ki) {
        // acc += c[ki-1]*Tcur ; Tprev = -Tprev  (becomes -T_{ki-2} init)
        fuse_k<<<ewb, 256, 0, stream>>>(ar, ai, tcr, tci, tpr, tpi, c, ki - 1, 1, NC);
        // Tprev_buf += -2*w*Tcur[col]  ->  T_ki = -2*A(T_{ki-1}) - T_{ki-2}
        edge_prop<<<eb, 256, 0, stream>>>(ei, ew, tcr, tci, tpr, tpi, -2.0f, E);
        float* t;
        t = tpr; tpr = tcr; tcr = t;
        t = tpi; tpi = tci; tci = t;
    }
    // acc += c5*T5
    fuse_k<<<ewb, 256, 0, stream>>>(ar, ai, tcr, tci, tpr, tpi, c, 5, 0, NC);

    int gb = (N / 16 + 7) / 8;   // 8 row-tiles per block
    gemm_silu<<<gb, 256, 0, stream>>>(ar, ai, Wr, Wi, (float*)d_out, N);
}
